// TransformerBlockQuantum_65481071410061
// MI455X (gfx1250) — matrix operand order repaired and checked by reading, not compiled, not run on hardware
//
#include <hip/hip_runtime.h>
#include <hip/hip_bf16.h>
#include <stdint.h>

// ---------------------------------------------------------------------------
// TransformerBlockQuantum for MI455X (gfx1250, wave32, WMMA)
//   B=4, S=2048, E=1024, H=16, DK=64, FFN=4096, NQ=8
//   attn (bf16 WMMA flash, 64-key blocks, DPP softmax reductions)
//   -> LN1 -> cos-feature up-proj
//   -> bf16 WMMA GEMM with TDM (tensor_load_to_lds) double-buffered staging
//   -> LN2
// ---------------------------------------------------------------------------

#define Bz   4
#define Sz   2048
#define Ez   1024
#define Hz   16
#define DKz  64
#define FFNz 4096
#define NQz  8
#define TOK  (Bz * Sz)   // 8192 tokens

typedef unsigned short ushort_t;
typedef __attribute__((ext_vector_type(16))) __bf16 v16bf;
typedef __attribute__((ext_vector_type(8)))  float  v8f;

union BF16x16 { v16bf v; uint4 q[2]; ushort_t s[16]; };
union U4x8    { uint4 q; ushort_t s[8]; };

__device__ __forceinline__ ushort_t f2bf_u16(float f) {
  union { float f; uint32_t u; } a; a.f = f;
  return (ushort_t)((a.u + 0x7FFFu + ((a.u >> 16) & 1u)) >> 16); // RNE
}

// --------------------------- cross-lane reductions -------------------------
// Row-of-16 butterfly: xor 1,2,4 via DPP8 (pure VALU), xor 8 via DPP16
// row_ror:8 (rotation by 8 within a 16-lane row == xor 8). No LDS, no waits.
constexpr int dpp8_sel_xor(int m) {
  int s = 0;
  for (int i = 0; i < 8; ++i) s |= ((i ^ m) & 7) << (3 * i);
  return s;
}
enum : int { SELX1 = dpp8_sel_xor(1), SELX2 = dpp8_sel_xor(2), SELX4 = dpp8_sel_xor(4) };

#if __has_builtin(__builtin_amdgcn_mov_dpp8) && __has_builtin(__builtin_amdgcn_update_dpp)
#define HAVE_DPP 1
#endif

__device__ __forceinline__ float red16_max(float v) {
#ifdef HAVE_DPP
  v = fmaxf(v, __int_as_float(__builtin_amdgcn_mov_dpp8(__float_as_int(v), SELX1)));
  v = fmaxf(v, __int_as_float(__builtin_amdgcn_mov_dpp8(__float_as_int(v), SELX2)));
  v = fmaxf(v, __int_as_float(__builtin_amdgcn_mov_dpp8(__float_as_int(v), SELX4)));
  v = fmaxf(v, __int_as_float(
          __builtin_amdgcn_update_dpp(0, __float_as_int(v), 0x128 /*row_ror:8*/,
                                      0xf, 0xf, true)));
#else
#pragma unroll
  for (int msk = 1; msk < 16; msk <<= 1) v = fmaxf(v, __shfl_xor(v, msk, 32));
#endif
  return v;
}

__device__ __forceinline__ float red16_sum(float v) {
#ifdef HAVE_DPP
  v += __int_as_float(__builtin_amdgcn_mov_dpp8(__float_as_int(v), SELX1));
  v += __int_as_float(__builtin_amdgcn_mov_dpp8(__float_as_int(v), SELX2));
  v += __int_as_float(__builtin_amdgcn_mov_dpp8(__float_as_int(v), SELX4));
  v += __int_as_float(
      __builtin_amdgcn_update_dpp(0, __float_as_int(v), 0x128, 0xf, 0xf, true));
#else
#pragma unroll
  for (int msk = 1; msk < 16; msk <<= 1) v += __shfl_xor(v, msk, 32);
#endif
  return v;
}

// ------------------------------ TDM support --------------------------------
#if __has_builtin(__builtin_amdgcn_tensor_load_to_lds) && \
    __has_builtin(__builtin_amdgcn_s_wait_tensorcnt)
#define HAVE_TDM 1
#endif

#ifdef HAVE_TDM
typedef unsigned int __attribute__((ext_vector_type(4))) v4u;
typedef int          __attribute__((ext_vector_type(8))) v8i;
typedef int          __attribute__((ext_vector_type(4))) v4i;

// 2D tile load: bf16 elements, tile (tile_d0 x tile_d1) from a row-major
// tensor with row stride `stride0` elements. LDS padding: +4 dwords after
// every 32 dwords -> padded row stride of 72 halves (matches LP).
// D# bitfields per CDNA5 ISA section 8.3/8.4.
// clang-23 builtin form (probe-confirmed 6 args):
//   (uint32x4 g0, int32x8 g1, int32x4 g2, int32x4 g3, int32x8 aux, i32 cpol)
__device__ __forceinline__ void tdm_load_tile_2d(
    uint32_t lds_addr, const ushort_t* gptr,
    uint32_t tile_d0, uint32_t tile_d1,
    uint32_t tensor_d0, uint32_t tensor_d1, uint32_t stride0) {
  const uint64_t ga = (uint64_t)(uintptr_t)gptr;
  v4u g0;
  g0[0] = 1u;                                      // count=1 (valid), user mode
  g0[1] = lds_addr;                                // lds_addr [63:32]
  g0[2] = (uint32_t)ga;                            // global_addr [95:64]
  g0[3] = (uint32_t)((ga >> 32) & 0x01FFFFFFu) |   // global_addr [120:96]
          (2u << 30);                              // type=2 ("image")
  v8i g1;
  g1[0] = (int)((1u << 16) |                       // data_size = 2 bytes
                (1u << 20) |                       // pad_enable
                (4u << 22) |                       // pad_interval: 32 dwords
                (3u << 25));                       // pad_amount: 4 dwords
  g1[1] = (int)(tensor_d0 << 16);                  // tensor_dim0[15:0] @ [63:48]
  g1[2] = (int)((tensor_d0 >> 16) | (tensor_d1 << 16)); // dim0 hi | dim1 lo
  g1[3] = (int)((tensor_d1 >> 16) | (tile_d0 << 16));   // dim1 hi | tile_dim0
  g1[4] = (int)tile_d1;                            // tile_dim1 (tile_dim2=0)
  g1[5] = (int)stride0;                            // tensor_dim0_stride lo
  g1[6] = 0;                                       // stride0 hi | dim1_stride lo
  g1[7] = 0;
  v4i g2 = { 1, 0, 0, 0 };                         // tensor_dim2=1 (benign 3D)
  v4i g3 = { 0, 0, 0, 0 };
  v8i aux = { 0, 0, 0, 0, 0, 0, 0, 0 };            // clang-23 extra operand
  __builtin_amdgcn_tensor_load_to_lds(g0, g1, g2, g3, aux, 0);
}
#endif

// ---------------------------------------------------------------------------
__global__ void __launch_bounds__(256)
k_f32_to_bf16(const float* __restrict__ in, ushort_t* __restrict__ out, int n) {
  int i = (blockIdx.x * 256 + threadIdx.x) * 4;
  if (i + 3 < n) {
    float4 v = *(const float4*)(in + i);
    ushort_t o4[4] = { f2bf_u16(v.x), f2bf_u16(v.y), f2bf_u16(v.z), f2bf_u16(v.w) };
    *(uint2*)(out + i) = *(const uint2*)o4;
  } else {
    for (; i < n; ++i) out[i] = f2bf_u16(in[i]);
  }
}

// ---------------------------------------------------------------------------
// Flash attention, Q=K=V = x[b,:,h*64:(h+1)*64].
// Block: 256 thr = 8 waves; wave handles 16 query rows; block = 128 rows.
// Key block = 64 keys -> 16 WMMA per softmax update. grid = (B*H, S/128)
// LDS rows padded to 72 halves (144 B) to spread banks.
// ---------------------------------------------------------------------------
#define LP 72  // padded LDS row stride in halves

__global__ void __launch_bounds__(256)
k_attention(const ushort_t* __restrict__ xbf, const float* __restrict__ x,
            float* __restrict__ y1) {
  const int bh   = blockIdx.x;
  const int b    = bh / Hz;
  const int hh   = bh % Hz;
  const int wave = threadIdx.x >> 5;
  const int lane = threadIdx.x & 31;
  const int lane15 = lane & 15;
  const int kg     = lane >> 4;          // lane-half selects K-group (ISA A/B layouts)
  const int q0   = blockIdx.y * 128 + wave * 16;
  const int col0 = hh * DKz;

  __shared__ __align__(16) ushort_t Kt[64 * LP];      // 64 keys x 64 dims (row major)
  __shared__ __align__(16) ushort_t Vt[64 * LP];      // transposed: dim x key
  __shared__ __align__(16) ushort_t Pst[8][16 * LP];  // per-wave P relayout tile

  // Q A-operands (16x32 bf16 each), resident in VGPRs for the whole kernel.
  // A layout: lanes 0-15 -> row=lane, K {0..7,16..23}; lanes 16-31 -> K {8..15,24..31}
  BF16x16 qa[2];
#pragma unroll
  for (int c = 0; c < 2; ++c) {
    const ushort_t* p =
        xbf + (size_t)((b * Sz + q0 + lane15) * Ez + col0 + c * 32 + kg * 8);
    qa[c].q[0] = *(const uint4*)p;
    qa[c].q[1] = *(const uint4*)(p + 16);
  }

  float m[8], l[8];
  v8f o[4] = {};
#pragma unroll
  for (int e = 0; e < 8; ++e) { m[e] = -3.0e38f; l[e] = 0.0f; }

  for (int j0 = 0; j0 < Sz; j0 += 64) {
    __syncthreads();
    { // stage 64x64 key tile (also serves as V); build transposed copy for P*V
      const int t   = threadIdx.x;
      const int r   = t >> 2;            // key row 0..63
      const int c16 = (t & 3) * 16;      // dim 0,16,32,48
      const ushort_t* p = xbf + (size_t)((b * Sz + j0 + r) * Ez + col0 + c16);
      U4x8 a0, a1;
      a0.q = *(const uint4*)p;
      a1.q = *(const uint4*)(p + 8);
      *(uint4*)&Kt[r * LP + c16]     = a0.q;
      *(uint4*)&Kt[r * LP + c16 + 8] = a1.q;
#pragma unroll
      for (int i = 0; i < 8; ++i) Vt[(c16 + i) * LP + r]     = a0.s[i];
#pragma unroll
      for (int i = 0; i < 8; ++i) Vt[(c16 + 8 + i) * LP + r] = a1.s[i];
      if (j0 + 64 < Sz) __builtin_prefetch(p + 64 * Ez, 0, 0); // next tile
    }
    __syncthreads();

    // scores S = Q K^T : 16 queries x 64 keys = 4 C tiles, K-dim 64 = 2 chunks
    v8f sc[4] = {};
#pragma unroll
    for (int c = 0; c < 2; ++c) {
      const int d0 = c * 32 + kg * 8;
#pragma unroll
      for (int t4 = 0; t4 < 4; ++t4) {
        // B layout (32x16): lanes 0-15 -> N=lane, K {0..7,16..23}; lanes 16-31 -> K {8..15,24..31}
        BF16x16 bk;
        const int key = t4 * 16 + lane15;
        bk.q[0] = *(const uint4*)&Kt[key * LP + d0];
        bk.q[1] = *(const uint4*)&Kt[key * LP + d0 + 16];
        sc[t4] = __builtin_amdgcn_wmma_f32_16x16x32_bf16(false, qa[c].v, false, bk.v,
                                                         (short)0, sc[t4], false, false);
      }
    }

    // online softmax in fp32. C layout: lanes 0-15 hold rows 0-7, lanes 16-31
    // rows 8-15; butterfly masks 1..8 stay inside each 16-lane half (DPP).
#pragma unroll
    for (int e = 0; e < 8; ++e) {
      float s0 = sc[0][e] * 0.125f, s1 = sc[1][e] * 0.125f;   // 1/sqrt(DK)
      float s2 = sc[2][e] * 0.125f, s3 = sc[3][e] * 0.125f;
      float mx = red16_max(fmaxf(fmaxf(s0, s1), fmaxf(s2, s3)));
      const float mn = fmaxf(m[e], mx);
      const float al = __expf(m[e] - mn);
      const float p0 = __expf(s0 - mn), p1 = __expf(s1 - mn);
      const float p2 = __expf(s2 - mn), p3 = __expf(s3 - mn);
      const float rs = red16_sum((p0 + p1) + (p2 + p3));
      l[e] = l[e] * al + rs;
      m[e] = mn;
#pragma unroll
      for (int n = 0; n < 4; ++n) o[n][e] *= al;
      const int pr = (kg * 8 + e) * LP + lane15;
      Pst[wave][pr]      = f2bf_u16(p0);
      Pst[wave][pr + 16] = f2bf_u16(p1);
      Pst[wave][pr + 32] = f2bf_u16(p2);
      Pst[wave][pr + 48] = f2bf_u16(p3);
    }

    // P: C layout -> A layout via per-wave LDS tile (DS ops in-order per wave),
    // then O += P * V over 2 key-chunks x 4 dim-tiles.
#pragma unroll
    for (int c = 0; c < 2; ++c) {
      BF16x16 pa;
      pa.q[0] = *(const uint4*)&Pst[wave][lane15 * LP + c * 32 + kg * 8];
      pa.q[1] = *(const uint4*)&Pst[wave][lane15 * LP + c * 32 + kg * 8 + 16];
#pragma unroll
      for (int n = 0; n < 4; ++n) {
        BF16x16 bv;
        const int dim = n * 16 + lane15;
        bv.q[0] = *(const uint4*)&Vt[dim * LP + c * 32 + kg * 8];
        bv.q[1] = *(const uint4*)&Vt[dim * LP + c * 32 + kg * 8 + 16];
        o[n] = __builtin_amdgcn_wmma_f32_16x16x32_bf16(false, pa.v, false, bv.v,
                                                       (short)0, o[n], false, false);
      }
    }
  }

  // epilogue: y1 = x + attn   (C layout: row = kg*8+e, col = n*16+lane15)
#pragma unroll
  for (int e = 0; e < 8; ++e) {
    const float invl = 1.0f / l[e];
    const int r = kg * 8 + e;
#pragma unroll
    for (int n = 0; n < 4; ++n) {
      const size_t gi = (size_t)((b * Sz + q0 + r) * Ez + col0 + n * 16 + lane15);
      y1[gi] = x[gi] + o[n][e] * invl;
    }
  }
}

// ---------------------------------------------------------------------------
// LayerNorm over E=1024; one block per token; optional residual + bias fused.
// ---------------------------------------------------------------------------
__global__ void __launch_bounds__(256)
k_layernorm(const float* __restrict__ a, const float* __restrict__ res,
            const float* __restrict__ bias,
            const float* __restrict__ g, const float* __restrict__ be,
            float* __restrict__ out) {
  const int tkn = blockIdx.x;
  const size_t base = (size_t)tkn * Ez;
  const int tid = threadIdx.x;
  __shared__ float r1[256], r2[256];
  float v[4], s = 0.f, s2 = 0.f;
#pragma unroll
  for (int i = 0; i < 4; ++i) {
    const int idx = tid + i * 256;
    float t = a[base + idx];
    if (res)  t += res[base + idx];
    if (bias) t += bias[idx];
    v[i] = t; s += t; s2 += t * t;
  }
  r1[tid] = s; r2[tid] = s2;
  __syncthreads();
  for (int off = 128; off > 0; off >>= 1) {
    if (tid < off) { r1[tid] += r1[tid + off]; r2[tid] += r2[tid + off]; }
    __syncthreads();
  }
  const float mu  = r1[0] * (1.0f / Ez);
  const float var = fmaxf(r2[0] * (1.0f / Ez) - mu * mu, 0.0f);
  const float rs  = rsqrtf(var + 1e-5f);
#pragma unroll
  for (int i = 0; i < 4; ++i) {
    const int idx = tid + i * 256;
    out[base + idx] = (v[i] - mu) * rs * g[idx] + be[idx];
  }
}

// ---------------------------------------------------------------------------
// hid[t,f] = relu( sum_q cos(h[t,q]) cos(theta[q]) * w1[f,q] + b1[f] ) -> bf16
// K=8: pure VALU (0.55 GFLOP total). One block per token, 16 outputs/thread.
// ---------------------------------------------------------------------------
__global__ void __launch_bounds__(256)
k_qffn_up(const float* __restrict__ h, const float* __restrict__ theta,
          const float* __restrict__ w1, const float* __restrict__ b1,
          ushort_t* __restrict__ hid) {
  const int tkn = blockIdx.x;
  __shared__ float qv[NQz];
  if (threadIdx.x < NQz)
    qv[threadIdx.x] = __cosf(h[(size_t)tkn * Ez + threadIdx.x]) * __cosf(theta[threadIdx.x]);
  __syncthreads();
  float q[NQz];
#pragma unroll
  for (int i = 0; i < NQz; ++i) q[i] = qv[i];
  const int f0 = threadIdx.x * 16;
#pragma unroll 4
  for (int f = f0; f < f0 + 16; ++f) {
    const float* wr = w1 + (size_t)f * NQz;
    float acc = b1[f];
#pragma unroll
    for (int i = 0; i < NQz; ++i) acc = fmaf(q[i], wr[i], acc);
    hid[(size_t)tkn * FFNz + f] = f2bf_u16(fmaxf(acc, 0.0f));
  }
}

// ---------------------------------------------------------------------------
// ffn[t,e] = sum_f hid[t,f] * w2[e,f]   (M=8192, N=1024, K=4096, bf16 WMMA)
// Block tile 64M x 128N, 8 waves each 16M x 64N, K step 64 -> 8 WMMA / stage.
// Double-buffered LDS staging via TDM: wave 0 issues tensor_load_to_lds for
// the next K-chunk while all waves compute on the current one; completion via
// s_wait_tensorcnt (TDM ops complete in-order per wave). TDM applies LDS
// padding (32 dwords data + 4 dwords pad) -> row stride 72 halves (= LP).
// grid (128, 8).
// ---------------------------------------------------------------------------
__global__ void __launch_bounds__(256)
k_ffn_gemm(const ushort_t* __restrict__ A, const ushort_t* __restrict__ Bw,
           float* __restrict__ C) {
  const int wave = threadIdx.x >> 5, lane = threadIdx.x & 31;
  const int lane15 = lane & 15, kg = lane >> 4;
  const int m0 = blockIdx.x * 64;
  const int n0 = blockIdx.y * 128;
  const int wm = (wave & 3) * 16;
  const int wn = (wave >> 2) * 64;
  __shared__ __align__(16) ushort_t As[2][64 * LP];
  __shared__ __align__(16) ushort_t Bs[2][128 * LP];
  v8f acc[4] = {};

#ifdef HAVE_TDM
  const uint32_t asAddr[2] = { (uint32_t)(uintptr_t)&As[0][0],
                               (uint32_t)(uintptr_t)&As[1][0] };
  const uint32_t bsAddr[2] = { (uint32_t)(uintptr_t)&Bs[0][0],
                               (uint32_t)(uintptr_t)&Bs[1][0] };
  if (wave == 0) { // pre-issue first tile pair
    tdm_load_tile_2d(asAddr[0], A  + (size_t)m0 * FFNz, 64, 64,  FFNz, TOK, FFNz);
    tdm_load_tile_2d(bsAddr[0], Bw + (size_t)n0 * FFNz, 64, 128, FFNz, Ez,  FFNz);
  }
#endif

  int buf = 0;
  for (int k0 = 0; k0 < FFNz; k0 += 64, buf ^= 1) {
#ifdef HAVE_TDM
    if (wave == 0) {
      if (k0 + 64 < FFNz) { // prefetch next tile pair into the other buffer
        tdm_load_tile_2d(asAddr[buf ^ 1], A  + (size_t)m0 * FFNz + (k0 + 64),
                         64, 64,  FFNz, TOK, FFNz);
        tdm_load_tile_2d(bsAddr[buf ^ 1], Bw + (size_t)n0 * FFNz + (k0 + 64),
                         64, 128, FFNz, Ez,  FFNz);
        __builtin_amdgcn_s_wait_tensorcnt(2); // current buffer's pair done
      } else {
        __builtin_amdgcn_s_wait_tensorcnt(0);
      }
    }
#else
    { // manual staging fallback
      const int t = threadIdx.x;
      const int ar = t >> 2, ac = (t & 3) * 16;
      const ushort_t* ap = &A[(size_t)(m0 + ar) * FFNz + k0 + ac];
      *(uint4*)&As[buf][ar * LP + ac]     = *(const uint4*)ap;
      *(uint4*)&As[buf][ar * LP + ac + 8] = *(const uint4*)(ap + 8);
      if (k0 + 64 < FFNz) __builtin_prefetch(ap + 64, 0, 0);
      const int br = t >> 1, bc = (t & 1) * 32;
      const ushort_t* bp = &Bw[(size_t)(n0 + br) * FFNz + k0 + bc];
#pragma unroll
      for (int i = 0; i < 4; ++i)
        *(uint4*)&Bs[buf][br * LP + bc + i * 8] = *(const uint4*)(bp + i * 8);
    }
#endif
    __syncthreads(); // current buffer ready for all waves

#pragma unroll
    for (int c = 0; c < 2; ++c) {
      BF16x16 av;
      av.q[0] = *(const uint4*)&As[buf][(wm + lane15) * LP + c * 32 + kg * 8];
      av.q[1] = *(const uint4*)&As[buf][(wm + lane15) * LP + c * 32 + kg * 8 + 16];
#pragma unroll
      for (int n = 0; n < 4; ++n) {
        BF16x16 bv;
        const int bn = wn + n * 16 + lane15;
        bv.q[0] = *(const uint4*)&Bs[buf][bn * LP + c * 32 + kg * 8];
        bv.q[1] = *(const uint4*)&Bs[buf][bn * LP + c * 32 + kg * 8 + 16];
        acc[n] = __builtin_amdgcn_wmma_f32_16x16x32_bf16(false, av.v, false, bv.v,
                                                         (short)0, acc[n], false, false);
      }
    }
    __syncthreads(); // all reads done before next TDM overwrites other buffer
  }

#pragma unroll
  for (int e = 0; e < 8; ++e) {
    const int r = kg * 8 + e;
#pragma unroll
    for (int n = 0; n < 4; ++n)
      C[(size_t)(m0 + wm + r) * Ez + n0 + wn + n * 16 + lane15] = acc[n][e];
  }
}

// ---------------------------------------------------------------------------
extern "C" void kernel_launch(void* const* d_in, const int* in_sizes, int n_in,
                              void* d_out, int out_size, void* d_ws, size_t ws_size,
                              hipStream_t stream) {
  const float* x     = (const float*)d_in[0];
  const float* theta = (const float*)d_in[1];
  const float* w1    = (const float*)d_in[2];
  const float* b1    = (const float*)d_in[3];
  const float* w2    = (const float*)d_in[4];
  const float* b2    = (const float*)d_in[5];
  const float* g1    = (const float*)d_in[6];
  const float* be1   = (const float*)d_in[7];
  const float* g2    = (const float*)d_in[8];
  const float* be2   = (const float*)d_in[9];
  float* out = (float*)d_out;

  char* ws = (char*)d_ws;
  // region layout (bytes):  [0,16M) xbf | [16M,48M) y1 | both dead -> [0,64M) hid
  //                         [64M,96M) h | [96M,104M) w2bf | [104M,136M) ffn
  ushort_t* xbf  = (ushort_t*)(ws + 0);
  float*    y1   = (float*)   (ws + (size_t)(16u << 20));
  ushort_t* hid  = (ushort_t*)(ws + 0);
  float*    h    = (float*)   (ws + (size_t)(64u << 20));
  ushort_t* w2bf = (ushort_t*)(ws + (size_t)(96u << 20));
  float*    ffn  = (float*)   (ws + (size_t)(104u << 20));

  const int NX  = Bz * Sz * Ez;   // 8388608
  const int NW2 = Ez * FFNz;      // 4194304

  k_f32_to_bf16<<<dim3(NX  / 1024), dim3(256), 0, stream>>>(x,  xbf,  NX);
  k_f32_to_bf16<<<dim3(NW2 / 1024), dim3(256), 0, stream>>>(w2, w2bf, NW2);
  k_attention  <<<dim3(Bz * Hz, Sz / 128), dim3(256), 0, stream>>>(xbf, x, y1);
  k_layernorm  <<<dim3(TOK), dim3(256), 0, stream>>>(y1, nullptr, nullptr, g1, be1, h);
  k_qffn_up    <<<dim3(TOK), dim3(256), 0, stream>>>(h, theta, w1, b1, hid);
  k_ffn_gemm   <<<dim3(TOK / 64, Ez / 128), dim3(256), 0, stream>>>(hid, w2bf, ffn);
  k_layernorm  <<<dim3(TOK), dim3(256), 0, stream>>>(h, ffn, b2, g2, be2, out);

  (void)in_sizes; (void)n_in; (void)out_size; (void)ws_size;
}
